// ConvNeXtBlock_59768764891908
// MI455X (gfx1250) — compile-verified, matrix operand order checked by reading, not run on hardware
//
#include <hip/hip_runtime.h>

// ---------------------------------------------------------------------------
// Problem constants (from reference: N=32, C=256, M=4, H=W=56)
// ---------------------------------------------------------------------------
constexpr int NB   = 32;
constexpr int C    = 256;
constexpr int CH   = 1024;        // 4*C
constexpr int H    = 56;
constexpr int W    = 56;
constexpr int HW   = H * W;       // 3136
constexpr int HP   = 58;          // padded (zero ring for the 3x3 halo)
constexpr int WP   = 58;
constexpr int MTOT = NB * HW;     // 100352 rows of the implicit GEMMs
constexpr int K1   = 9 * C;       // 2304 (3x3 conv reduction)
constexpr int A0P_ELTS = NB * HP * WP * C;   // padded activation elements

typedef __attribute__((ext_vector_type(16))) _Float16 v16h;
typedef __attribute__((ext_vector_type(8)))  float    v8f;
typedef __attribute__((__vector_size__(16))) int      vi4;   // b128 payload type

#define AS1 __attribute__((address_space(1)))
#define AS3 __attribute__((address_space(3)))

// --- CDNA5 async global->LDS path (probe-confirmed present on this toolchain)
#if defined(__HIP_DEVICE_COMPILE__)
#  if __has_builtin(__builtin_amdgcn_global_load_async_to_lds_b128)
#    define HAS_ASYNC_LDS 1
#  else
#    define HAS_ASYNC_LDS 0
#  endif
#  if __has_builtin(__builtin_amdgcn_s_wait_asynccnt)
#    define WAIT_ASYNC(n) __builtin_amdgcn_s_wait_asynccnt(n)
#  else
#    define WAIT_ASYNC(n) asm volatile("s_wait_asynccnt " #n ::: "memory")
#  endif
#else
#  define HAS_ASYNC_LDS 0
#  define WAIT_ASYNC(n)
#endif

__device__ __forceinline__ float hsign(float x) {
    return x > 0.0f ? 1.0f : (x < 0.0f ? -1.0f : 0.0f);
}
__device__ __forceinline__ float clamp11(float x) {
    return fminf(1.0f, fmaxf(-1.0f, x));
}

union FragH { v16h h; uint4 u[2]; };

// ---------------------------------------------------------------------------
// zero the padded activation buffer (border must be 0 every call; the harness
// poisons the workspace once and never re-poisons, so do it deterministically)
// ---------------------------------------------------------------------------
__global__ __launch_bounds__(256) void zero_u4(uint4* __restrict__ p)
{
    uint4 z; z.x = z.y = z.z = z.w = 0u;
    p[(size_t)blockIdx.x * 256 + threadIdx.x] = z;
}

// ---------------------------------------------------------------------------
// hardsign(x): NCHW fp32 -> padded NHWC fp16 via LDS 32x32 transpose.
// Reads fully coalesced (lane -> consecutive spatial), writes 64B chunks
// (lane -> consecutive channel).
// ---------------------------------------------------------------------------
__global__ __launch_bounds__(256) void sign_transpose(
    const float* __restrict__ x, _Float16* __restrict__ a0p)
{
    __shared__ float t[32][33];                    // +1 pad: conflict-free
    const int blk  = blockIdx.x;                   // n*(98*8) + spT*8 + cT
    const int cT   = blk & 7;
    const int tmp  = blk >> 3;
    const int spT  = tmp % (HW / 32);              // 98 spatial tiles
    const int n    = tmp / (HW / 32);
    const int lane = threadIdx.x & 31;
    const int grp  = threadIdx.x >> 5;             // 0..7

#pragma unroll
    for (int p = 0; p < 4; ++p) {
        const int cl = grp + p * 8;                // channel-local 0..31
        const int c  = cT * 32 + cl;
        const int sp = spT * 32 + lane;
        t[cl][lane] = hsign(x[(size_t)(n * C + c) * HW + sp]);
    }
    __syncthreads();
#pragma unroll
    for (int p = 0; p < 4; ++p) {
        const int spl = grp + p * 8;
        const int sp  = spT * 32 + spl;
        const int h   = sp / W, w = sp - h * W;
        const int c   = cT * 32 + lane;
        a0p[((size_t)((n * HP + h + 1) * WP) + (w + 1)) * C + c] =
            (_Float16)t[lane][spl];
    }
}

// ---------------------------------------------------------------------------
// weight prep: clamp(-1,1) -> fp16, relayout w_dw (C,C,3,3) -> [cout][k]
// with k = (ky*3+kx)*C + cin so each 32-wide K tile is contiguous.
// ---------------------------------------------------------------------------
__global__ __launch_bounds__(256) void wprep_dw(
    const float* __restrict__ w, _Float16* __restrict__ o)
{
    int i    = blockIdx.x * 256 + threadIdx.x;     // < C*K1 = 589824
    int cout = i / K1;
    int k    = i - cout * K1;
    int slab = k >> 8;                             // ky*3+kx
    int cin  = k & (C - 1);
    int ky   = slab / 3, kx = slab - 3 * (slab / 3);
    o[i] = (_Float16)clamp11(w[((cout * C + cin) * 3 + ky) * 3 + kx]);
}

__global__ __launch_bounds__(256) void wprep_clamp(
    const float* __restrict__ w, _Float16* __restrict__ o, int n)
{
    int i = blockIdx.x * 256 + threadIdx.x;
    if (i < n) o[i] = (_Float16)clamp11(w[i]);
}

// ---------------------------------------------------------------------------
// LayerNorm over channels (256) + hardsign, one wave (32 lanes) per row.
// jnp.var is the population variance -> E[x^2] - E[x]^2.
// ---------------------------------------------------------------------------
__global__ __launch_bounds__(256) void ln_sign_kernel(
    const float* __restrict__ y, const float* __restrict__ g,
    const float* __restrict__ b, _Float16* __restrict__ o)
{
    const int row  = blockIdx.x * 8 + (threadIdx.x >> 5);
    const int lane = threadIdx.x & 31;
    const float4* yr = (const float4*)(y + (size_t)row * C);
    float4 v0 = yr[lane * 2];
    float4 v1 = yr[lane * 2 + 1];
    float e[8] = {v0.x, v0.y, v0.z, v0.w, v1.x, v1.y, v1.z, v1.w};

    float s = 0.f, q = 0.f;
#pragma unroll
    for (int i = 0; i < 8; ++i) { s += e[i]; q += e[i] * e[i]; }
#pragma unroll
    for (int off = 16; off > 0; off >>= 1) {
        s += __shfl_xor(s, off, 32);
        q += __shfl_xor(q, off, 32);
    }
    float mean = s * (1.0f / 256.0f);
    float var  = q * (1.0f / 256.0f) - mean * mean;
    float rs   = rsqrtf(var + 1e-5f);

    int c0 = lane * 8;
    union { _Float16 h[8]; uint4 u; } pk;
#pragma unroll
    for (int i = 0; i < 8; ++i) {
        float xn = (e[i] - mean) * rs * g[c0 + i] + b[c0 + i];
        pk.h[i] = (_Float16)hsign(xn);
    }
    ((uint4*)(o + (size_t)row * C))[lane] = pk.u;
}

// ---------------------------------------------------------------------------
// WMMA GEMM: D = A(MTOT x KTOT, fp16) * B(KTOT x NOUT, fp16 as [cout][k])
// Block tile 64x256, BK=32. 8 waves (wave32): 2x4 wave grid, each wave owns
// a 32x64 tile = 2x4 accumulators of v_wmma_f32_16x16x32_f16.
//
// Triple-buffered async pipeline, ONE barrier per K-step:
//   issue(k+1) -> s_wait_asynccnt 5 (drains batch k; 5 async instrs/batch,
//   in-order completion) -> barrier -> compute from buf[k%3].
// Buffer b is re-written (for step k+3) only after bar(k+1), which fences all
// waves' LDS reads of b (step k) -> race-free with a single barrier.
//
// IS3X3: implicit im2col over the ZERO-PADDED 58x58 NHWC activations, so all
// loads are in-bounds: uniform EXEC (WMMA requirement) and exact async counts.
// MODE 0: store fp32 (pre-LN).  MODE 1: gelu+hardsign -> fp16.
// MODE 2: out = x + gamma*acc, vectorized float4 residual to NCHW.
// ---------------------------------------------------------------------------
template<int KTOT, int NOUT, int MODE, bool IS3X3>
__global__ __launch_bounds__(256) void gemm_ws(
    const _Float16* __restrict__ A, const _Float16* __restrict__ Bm,
    float* __restrict__ outF, _Float16* __restrict__ outH,
    const float* __restrict__ xres, const float* __restrict__ gammaP)
{
    __shared__ __align__(16) _Float16 Alds[3][64 * 32];
    __shared__ __align__(16) _Float16 Blds[3][256 * 32];

    const int tid  = threadIdx.x;
    const int arow = tid >> 2;           // 0..63   (A tile row this thread fills)
    const int aseg = tid & 3;            // 0..3    (16B segment within 32 halves)
    const int m    = blockIdx.x * 64 + arow;
    int an = 0, ah = 0, aw = 0;
    if (IS3X3) { an = m / HW; int rem = m - an * HW; ah = rem / W; aw = rem - ah * W; }

    const int wave = tid >> 5, lane = tid & 31;
    const int wm = wave >> 2, wn = wave & 3;     // 2 x 4 wave grid
    const int fr = lane & 15, fhi = lane >> 4;   // fragment row/col + K-half select

    // issue one K-step's tiles (5 async b128 per thread: 1 A + 4 B)
    auto issue = [&](int kt, int buf) {
        size_t gidx;
        if (IS3X3) {
            int slab = kt >> 3;                       // (ky,kx): 8 k-tiles per slab
            int ky = slab / 3, kx = slab - 3 * (slab / 3);
            int cb = ((kt & 7) << 5) + aseg * 8;
            gidx = ((size_t)((an * HP + ah + ky) * WP + aw + kx) << 8) + (size_t)cb;
        } else {
            gidx = (size_t)m * KTOT + (size_t)(kt * 32 + aseg * 8);
        }
        _Float16* adst = &Alds[buf][arow * 32 + aseg * 8];
#if HAS_ASYNC_LDS
        __builtin_amdgcn_global_load_async_to_lds_b128(
            (AS1 vi4*)(A + gidx), (AS3 vi4*)adst, 0, 0);
#else
        *(uint4*)adst = *(const uint4*)(A + gidx);
#endif
#pragma unroll
        for (int cch = 0; cch < 4; ++cch) {
            int lin = tid + cch * 256;
            int br = lin >> 2, bsg = lin & 3;
            size_t gb = (size_t)(blockIdx.y * 256 + br) * KTOT
                      + (size_t)(kt * 32 + bsg * 8);
            _Float16* bdst = &Blds[buf][br * 32 + bsg * 8];
#if HAS_ASYNC_LDS
            __builtin_amdgcn_global_load_async_to_lds_b128(
                (AS1 vi4*)(Bm + gb), (AS3 vi4*)bdst, 0, 0);
#else
            *(uint4*)bdst = *(const uint4*)(Bm + gb);
#endif
        }
    };

    v8f zero = {};
    v8f acc[2][4];
#pragma unroll
    for (int i = 0; i < 2; ++i)
#pragma unroll
        for (int j = 0; j < 4; ++j) acc[i][j] = zero;

    const int nkt = KTOT / 32;
    issue(0, 0);
    for (int kt = 0; kt < nkt; ++kt) {
        const int cur = kt % 3;
        if (kt + 1 < nkt) {
            issue(kt + 1, (kt + 1) % 3);
            WAIT_ASYNC(5);     // batch kt landed (in-order), batch kt+1 in flight
        } else {
            WAIT_ASYNC(0);
        }
        __syncthreads();

        // fragments per ISA 16-bit layout:
        // lane<16: K 0-7 (v0-3) + 16-23 (v4-7); lane>=16: K 8-15 + 24-31
        FragH fa[2], fb[4];
#pragma unroll
        for (int i = 0; i < 2; ++i) {
            int base = (wm * 32 + i * 16 + fr) * 32 + fhi * 8;
            fa[i].u[0] = *(const uint4*)&Alds[cur][base];
            fa[i].u[1] = *(const uint4*)&Alds[cur][base + 16];
        }
#pragma unroll
        for (int j = 0; j < 4; ++j) {
            int base = (wn * 64 + j * 16 + fr) * 32 + fhi * 8;
            fb[j].u[0] = *(const uint4*)&Blds[cur][base];
            fb[j].u[1] = *(const uint4*)&Blds[cur][base + 16];
        }
#pragma unroll
        for (int i = 0; i < 2; ++i)
#pragma unroll
            for (int j = 0; j < 4; ++j)
                acc[i][j] = __builtin_amdgcn_wmma_f32_16x16x32_f16(
                    false, fa[i].h, false, fb[j].h,
                    (short)0, acc[i][j], false, false);
    }

    // ---- epilogue: C/D layout: lane holds col (lane&15), rows v + 8*(lane>>4)
    const int rowBase = blockIdx.x * 64 + wm * 32;
    const int colBase = blockIdx.y * 256 + wn * 64;
    float gam = 0.0f;
    if constexpr (MODE == 2) gam = *gammaP;

#pragma unroll
    for (int i = 0; i < 2; ++i) {
#pragma unroll
        for (int j = 0; j < 4; ++j) {
            const int col = colBase + j * 16 + fr;
            if constexpr (MODE == 2) {
                // 8 accumulator rows = 8 consecutive NCHW spatial positions
                const int rr0 = rowBase + i * 16 + fhi * 8;
                const int n   = rr0 / HW;
                const int sp0 = rr0 - n * HW;
                const size_t base = (size_t)(n * C + col) * HW + sp0;
                float4 x0 = *(const float4*)(xres + base);
                float4 x1 = *(const float4*)(xres + base + 4);
                float4 r0, r1;
                r0.x = x0.x + gam * acc[i][j][0];
                r0.y = x0.y + gam * acc[i][j][1];
                r0.z = x0.z + gam * acc[i][j][2];
                r0.w = x0.w + gam * acc[i][j][3];
                r1.x = x1.x + gam * acc[i][j][4];
                r1.y = x1.y + gam * acc[i][j][5];
                r1.z = x1.z + gam * acc[i][j][6];
                r1.w = x1.w + gam * acc[i][j][7];
                *(float4*)(outF + base)     = r0;
                *(float4*)(outF + base + 4) = r1;
            } else {
#pragma unroll
                for (int v = 0; v < 8; ++v) {
                    const int rr = rowBase + i * 16 + fhi * 8 + v;
                    float val = acc[i][j][v];
                    if constexpr (MODE == 0) {
                        outF[(size_t)rr * NOUT + col] = val;
                    } else {
                        float gl = 0.5f * val *
                                   (1.0f + erff(val * 0.70710678118654752f));
                        outH[(size_t)rr * NOUT + col] = (_Float16)hsign(gl);
                    }
                }
            }
        }
    }
}

// ---------------------------------------------------------------------------
// Launch
// ---------------------------------------------------------------------------
extern "C" void kernel_launch(void* const* d_in, const int* in_sizes, int n_in,
                              void* d_out, int out_size, void* d_ws, size_t ws_size,
                              hipStream_t stream)
{
    (void)in_sizes; (void)n_in; (void)out_size; (void)ws_size;
    const float* x     = (const float*)d_in[0];
    const float* w_dw  = (const float*)d_in[1];
    const float* w_pw1 = (const float*)d_in[2];
    const float* w_pw2 = (const float*)d_in[3];
    const float* ln_g  = (const float*)d_in[4];
    const float* ln_b  = (const float*)d_in[5];
    const float* gamma = (const float*)d_in[6];
    float* out = (float*)d_out;

    char* ws = (char*)d_ws;
    size_t off = 0;
    auto alloc = [&](size_t bytes) -> void* {
        void* p = ws + off;
        off += (bytes + 255) & ~(size_t)255;
        return p;
    };
    _Float16* a0p = (_Float16*)alloc((size_t)A0P_ELTS * 2);   // padded sign(x)
    float*    y1  = (float*)   alloc((size_t)MTOT * C  * 4);  // conv1 out fp32
    _Float16* a1  = (_Float16*)alloc((size_t)MTOT * C  * 2);  // LN+sign fp16
    _Float16* a2  = (_Float16*)alloc((size_t)MTOT * CH * 2);  // gelu+sign fp16
    _Float16* wB1 = (_Float16*)alloc((size_t)C  * K1 * 2);    // 3x3 weights
    _Float16* wB2 = (_Float16*)alloc((size_t)CH * C  * 2);    // pw1 weights
    _Float16* wB3 = (_Float16*)alloc((size_t)C  * CH * 2);    // pw2 weights

    // zero padded activation buffer (border ring), then fill interior
    zero_u4<<<(A0P_ELTS * 2 / 16) / 256, 256, 0, stream>>>((uint4*)a0p);
    sign_transpose<<<NB * (HW / 32) * (C / 32), 256, 0, stream>>>(x, a0p);
    wprep_dw<<<(C * K1) / 256, 256, 0, stream>>>(w_dw, wB1);
    wprep_clamp<<<(CH * C) / 256, 256, 0, stream>>>(w_pw1, wB2, CH * C);
    wprep_clamp<<<(C * CH) / 256, 256, 0, stream>>>(w_pw2, wB3, C * CH);

    // conv1: 3x3 implicit GEMM, K=2304, N=256
    gemm_ws<K1, C, 0, true><<<dim3(MTOT / 64, 1), 256, 0, stream>>>(
        a0p, wB1, y1, nullptr, nullptr, nullptr);
    // LayerNorm(channels) + hardsign
    ln_sign_kernel<<<MTOT / 8, 256, 0, stream>>>(y1, ln_g, ln_b, a1);
    // conv2: 1x1 expand 256->1024, fused gelu+hardsign epilogue
    gemm_ws<C, CH, 1, false><<<dim3(MTOT / 64, CH / 256), 256, 0, stream>>>(
        a1, wB2, nullptr, a2, nullptr, nullptr);
    // conv3: 1x1 project 1024->256, fused residual (x + gamma*h) to NCHW
    gemm_ws<CH, C, 2, false><<<dim3(MTOT / 64, 1), 256, 0, stream>>>(
        a2, wB3, out, nullptr, x, gamma);
}
